// Flow_48619029791058
// MI455X (gfx1250) — matrix operand order repaired and checked by reading, not compiled, not run on hardware
//
#include <hip/hip_runtime.h>

typedef _Float16 half_t;
typedef __attribute__((ext_vector_type(16))) _Float16 v16h;
typedef __attribute__((ext_vector_type(8)))  _Float16 v8h;
typedef __attribute__((ext_vector_type(8)))  float    v8f;

#define B_SZ     4096
#define NC       64
#define NT       16
#define H_DIM    256
#define HID      256
#define K_MIX    20
#define IN_GRU   209
#define CBM      208
#define CBM_PAD  224
#define G3H      768
#define MLP_IN   464
#define MLP_INP  480
#define P_OUT    60
#define P_PAD    64
#define MBLK     32      // GRU rows per WG
#define MBLK_M   16      // MLP rows per block (double-buffered)
#define HALF_L2PI 0.9189385332046727f

#define WMMA16(a, b, c) \
  __builtin_amdgcn_wmma_f32_16x16x32_f16(false, (a), false, (b), (short)0, (c), false, false)

// ---------------- CDNA5 async LDS<->global helpers ----------------
// LDS operand = wave-relative LDS byte offset = low 32 bits of the generic
// shared-aperture address.

__device__ inline void async_ld128(unsigned ldsOff, unsigned long long ga) {
  asm volatile("global_load_async_to_lds_b128 %0, %1, off"
               :: "v"(ldsOff), "v"(ga) : "memory");
}
__device__ inline void async_st128(unsigned long long ga, unsigned ldsOff) {
  asm volatile("global_store_async_from_lds_b128 %0, %1, off"
               :: "v"(ga), "v"(ldsOff) : "memory");
}
__device__ inline void wait_async0() {
  asm volatile("s_wait_asynccnt 0" ::: "memory");
}

// ---------------- WMMA fragment helpers (wave32, 16x16x32 f16) ----------------

__device__ inline v16h load_b_frag(const half_t* __restrict__ W, int rowStride,
                                   int nBase, int kc) {
  const int lane = threadIdx.x & 31;
  const int col  = nBase + (lane & 15);
  const int kb   = kc * 32 + (((lane >> 4) & 1) ? 16 : 0);
  return *(const v16h*)(W + (size_t)col * rowStride + kb);
}

__device__ inline v16h load_a_frag(const half_t* A, int rowStride, int kc) {
  const int lane = threadIdx.x & 31;
  const int m    = lane & 15;
  const int koff = kc * 32 + (((lane >> 4) & 1) ? 16 : 0);
  const half_t* p = A + (size_t)m * rowStride + koff;
  v8h lo = *(const v8h*)(p);
  v8h hi = *(const v8h*)(p + 8);
  v16h r;
#pragma unroll
  for (int i = 0; i < 8; ++i) { r[i] = lo[i]; r[i + 8] = hi[i]; }
  return r;
}

__device__ inline float sigmoidf_(float x) { return 1.0f / (1.0f + __expf(-x)); }

// ---------------- prep kernels ----------------

__global__ void prep_rows(const float* __restrict__ c, const float* __restrict__ b,
                          const float* __restrict__ m,
                          half_t* __restrict__ cbmh, int* __restrict__ count) {
  const int row = blockIdx.x;
  const int t   = threadIdx.x;
  if (t < CBM_PAD) {
    float v;
    if (t < NC + NT)            v = c[row * (NC + NT) + t];
    else if (t < NC + NT + NC)  v = b[row * NC + (t - (NC + NT))];
    else if (t < CBM)           v = m[row * NC + (t - (NC + NT + NC))];
    else                        v = 0.0f;
    cbmh[row * CBM_PAD + t] = (half_t)v;
  }
  if (t == 0) {
    int cnt = 0;
    for (int j = 0; j < NC; ++j)
      cnt += (m[row * NC + j] * (1.0f - b[row * NC + j]) > 0.5f) ? 1 : 0;
    count[row] = cnt;
  }
}

__global__ void prep_weights(const float* __restrict__ W_ih, const float* __restrict__ W_hh,
                             const float* __restrict__ W1,   const float* __restrict__ W2,
                             const float* __restrict__ W3,
                             half_t* __restrict__ Wih_h, half_t* __restrict__ Whh_h,
                             half_t* __restrict__ W1h,   half_t* __restrict__ W2h,
                             half_t* __restrict__ W3h) {
  const int i = blockIdx.x * blockDim.x + threadIdx.x;
  if (i < G3H * CBM_PAD) {
    int n = i / CBM_PAD, k = i % CBM_PAD;
    Wih_h[i] = (half_t)((k < CBM) ? W_ih[n * IN_GRU + 1 + k] : 0.0f);
  }
  if (i < G3H * H_DIM) Whh_h[i] = (half_t)W_hh[i];
  if (i < HID * MLP_INP) {
    int n = i / MLP_INP, k = i % MLP_INP;
    W1h[i] = (half_t)((k < MLP_IN) ? W1[n * MLP_IN + k] : 0.0f);
  }
  if (i < HID * HID) W2h[i] = (half_t)W2[i];
  if (i < P_PAD * H_DIM) {
    int n = i / H_DIM, k = i % H_DIM;
    W3h[i] = (half_t)((n < P_OUT) ? W3[n * H_DIM + k] : 0.0f);
  }
}

// ---------------- gi = cbm @ W_ih[:,1:]^T + b_ih   [4096 x 768] ----------------

__global__ void __launch_bounds__(32)
gi_gemm(const half_t* __restrict__ cbmh, const half_t* __restrict__ Wih,
        const float* __restrict__ b_ih, float* __restrict__ gi) {
  const int tileN = blockIdx.x % (G3H / 16);
  const int tileM = blockIdx.x / (G3H / 16);
  v8f cacc = {};
#pragma unroll
  for (int kc = 0; kc < CBM_PAD / 32; ++kc) {
    v16h a = load_a_frag(cbmh + (size_t)tileM * 16 * CBM_PAD, CBM_PAD, kc);
    v16h b = load_b_frag(Wih, CBM_PAD, tileN * 16, kc);
    cacc = WMMA16(a, b, cacc);
  }
  const int lane = threadIdx.x & 31;
  const int mAdd = ((lane >> 4) & 1) ? 8 : 0;
  const int col  = tileN * 16 + (lane & 15);
#pragma unroll
  for (int v = 0; v < 8; ++v)
    gi[(size_t)(tileM * 16 + v + mAdd) * G3H + col] = cacc[v] + b_ih[col];
}

// ---------------- GRU phase: 32 rows/WG, 16 waves, W_hh resident in VGPRs ----
// hs writeback via async LDS->global b128 stores issued by wave 0, fenced one
// step later with s_wait_asynccnt before hH is overwritten.

__global__ void __launch_bounds__(512)
gru_persist(const float* __restrict__ z,    const float* __restrict__ gi,
            const float* __restrict__ b_hh, const float* __restrict__ W_ih,
            const half_t* __restrict__ Whh, half_t* __restrict__ hs) {
  __shared__ __align__(16) half_t hH[MBLK][H_DIM];   // 16 KB
  __shared__ float zLds[MBLK][NC];                   // 8 KB

  const int tid  = threadIdx.x;
  const int wave = tid >> 5;            // 0..15  == j-tile
  const int lane = tid & 31;
  const int rowBase = blockIdx.x * MBLK;
  const int jt   = wave;
  const int colN = jt * 16 + (lane & 15);
  const int mAdd = ((lane >> 4) & 1) ? 8 : 0;

  for (int e = tid; e < MBLK * H_DIM; e += 512)
    hH[e >> 8][e & 255] = (half_t)0.0f;
  for (int e = tid; e < MBLK * NC; e += 512)
    zLds[e >> 6][e & 63] = z[(size_t)(rowBase + (e >> 6)) * NC + (e & 63)];

  // persistent W_hh fragments (r/u/n gates for this j-tile): 24 frags = 192 VGPR
  v16h BR[8], BU[8], BN[8];
#pragma unroll
  for (int kc = 0; kc < 8; ++kc) {
    BR[kc] = load_b_frag(Whh, H_DIM, jt * 16,       kc);
    BU[kc] = load_b_frag(Whh, H_DIM, jt * 16 + 256, kc);
    BN[kc] = load_b_frag(Whh, H_DIM, jt * 16 + 512, kc);
  }

  const float w0r = W_ih[(colN)       * IN_GRU];
  const float w0u = W_ih[(256 + colN) * IN_GRU];
  const float w0n = W_ih[(512 + colN) * IN_GRU];
  const float bhn = b_hh[512 + colN];

  float giR[2][8], giU[2][8], giN[2][8], hPrev[2][8];
#pragma unroll
  for (int mt = 0; mt < 2; ++mt)
#pragma unroll
    for (int v = 0; v < 8; ++v) {
      const int grow = rowBase + mt * 16 + v + mAdd;
      giR[mt][v] = gi[(size_t)grow * G3H + colN]       + b_hh[colN];
      giU[mt][v] = gi[(size_t)grow * G3H + 256 + colN] + b_hh[256 + colN];
      giN[mt][v] = gi[(size_t)grow * G3H + 512 + colN];
      hPrev[mt][v] = 0.0f;
    }
  __syncthreads();

  for (int t = 0; t < NC; ++t) {
    v8f accR[2] = {}, accU[2] = {}, accN[2] = {};
#pragma unroll
    for (int kc = 0; kc < 8; ++kc) {
#pragma unroll
      for (int mt = 0; mt < 2; ++mt) {
        v16h a = load_a_frag(&hH[mt * 16][0], H_DIM, kc);
        accR[mt] = WMMA16(a, BR[kc], accR[mt]);
        accU[mt] = WMMA16(a, BU[kc], accU[mt]);
        accN[mt] = WMMA16(a, BN[kc], accN[mt]);
      }
    }
    if (wave == 0) wait_async0();   // prior step's hs stores done reading hH
    __syncthreads();                // + all waves done reading hH for wmma
#pragma unroll
    for (int mt = 0; mt < 2; ++mt) {
#pragma unroll
      for (int v = 0; v < 8; ++v) {
        const int row  = mt * 16 + v + mAdd;
        const float zp = (t == 0) ? -1.0f : zLds[row][t - 1];
        const float r  = sigmoidf_(giR[mt][v] + zp * w0r + accR[mt][v]);
        const float u  = sigmoidf_(giU[mt][v] + zp * w0u + accU[mt][v]);
        const float n  = tanhf(giN[mt][v] + zp * w0n + r * (accN[mt][v] + bhn));
        const float hN = (1.0f - u) * n + u * hPrev[mt][v];
        hPrev[mt][v] = hN;
        hH[row][colN] = (half_t)hN;
      }
    }
    __syncthreads();                // hH complete for this step
    if (wave == 0) {                // async coalesced writeback: 32 x 512B rows
      for (int r = 0; r < MBLK; ++r) {
        unsigned lo = (unsigned)(uintptr_t)(&hH[r][0]) + lane * 16;
        unsigned long long ga =
            (unsigned long long)(uintptr_t)(hs + ((size_t)(rowBase + r) * NC + t) * H_DIM)
            + (unsigned)(lane * 16);
        async_st128(ga, lo);
      }
    }
  }
  // S_ENDPGM implies wait-idle: outstanding async stores drain before exit.
}

// ---------------- MLP phase: 16-row blocks, double-buffered async feat ------

__device__ inline void prefetch_feat(half_t* buf, const half_t* hsRow,
                                     const half_t* cbmRow, int lane) {
  // h part: 16 rows x 512 B
  for (int r = 0; r < MBLK_M; ++r) {
    unsigned lo = (unsigned)(uintptr_t)(buf + r * MLP_INP) + lane * 16;
    unsigned long long ga =
        (unsigned long long)(uintptr_t)(hsRow + (size_t)r * H_DIM) + (unsigned)(lane * 16);
    async_ld128(lo, ga);
  }
  // cbm part: 448 B per row (lanes 0..27), same source for all rows
  if (lane < 28) {
    unsigned long long ga = (unsigned long long)(uintptr_t)cbmRow + (unsigned)(lane * 16);
    for (int r = 0; r < MBLK_M; ++r) {
      unsigned lo = (unsigned)(uintptr_t)(buf + r * MLP_INP + H_DIM) + lane * 16;
      async_ld128(lo, ga);
    }
  }
}

__global__ void __launch_bounds__(256)
mlp_persist(const half_t* __restrict__ hs, const half_t* __restrict__ cbmh,
            const float* __restrict__ z,   const int* __restrict__ count,
            const half_t* __restrict__ W1h, const float* __restrict__ b1,
            const half_t* __restrict__ W2h, const float* __restrict__ b2,
            const half_t* __restrict__ W3h, const float* __restrict__ b3,
            float* __restrict__ llbuf) {
  __shared__ __align__(16) half_t featA[MBLK_M][MLP_INP];   // 15 KB
  __shared__ __align__(16) half_t featB[MBLK_M][MLP_INP];   // 15 KB
  __shared__ __align__(16) half_t h1H[MBLK_M][HID];         // 8 KB
  __shared__ __align__(16) half_t h2H[MBLK_M][HID];         // 8 KB
  __shared__ __align__(16) float  pLds[MBLK_M][P_PAD];      // 4 KB

  const int tid  = threadIdx.x;
  const int wave = tid >> 5;                 // 0..7
  const int lane = tid & 31;
  const int mAdd = ((lane >> 4) & 1) ? 8 : 0;
  const int nt0 = wave, nt1 = wave + 8;

  // persistent weights: W1 30 frags, W2 16 frags, W3 8 frags
  v16h W1f[2][15], W2f[2][8], W3f[8];
#pragma unroll
  for (int kc = 0; kc < 15; ++kc) {
    W1f[0][kc] = load_b_frag(W1h, MLP_INP, nt0 * 16, kc);
    W1f[1][kc] = load_b_frag(W1h, MLP_INP, nt1 * 16, kc);
  }
#pragma unroll
  for (int kc = 0; kc < 8; ++kc) {
    W2f[0][kc] = load_b_frag(W2h, HID, nt0 * 16, kc);
    W2f[1][kc] = load_b_frag(W2h, HID, nt1 * 16, kc);
    W3f[kc]    = load_b_frag(W3h, H_DIM, (wave & 3) * 16, kc);
  }
  const float b1v0 = b1[nt0 * 16 + (lane & 15)];
  const float b1v1 = b1[nt1 * 16 + (lane & 15)];
  const float b2v0 = b2[nt0 * 16 + (lane & 15)];
  const float b2v1 = b2[nt1 * 16 + (lane & 15)];
  const int   col3 = (wave & 3) * 16 + (lane & 15);
  const float b3v  = (col3 < P_OUT) ? b3[col3] : 0.0f;

  const int totalBlk = (B_SZ * NC) / MBLK_M;   // 16384
  // prologue prefetch into featA
  if (wave == 0 && blockIdx.x < totalBlk) {
    const int rb = blockIdx.x * MBLK_M;
    prefetch_feat(&featA[0][0], hs + (size_t)rb * H_DIM,
                  cbmh + (size_t)(rb / NC) * CBM_PAD, lane);
  }

  int iter = 0;
  for (int blk = blockIdx.x; blk < totalBlk; blk += gridDim.x, ++iter) {
    half_t* cur = (iter & 1) ? &featB[0][0] : &featA[0][0];
    half_t* nxt = (iter & 1) ? &featA[0][0] : &featB[0][0];
    const int rowBase = blk * MBLK_M;

    if (wave == 0) wait_async0();   // cur buffer is resident
    __syncthreads();

    const int nblk = blk + gridDim.x;
    if (wave == 0 && nblk < totalBlk) {
      const int rb = nblk * MBLK_M;
      prefetch_feat(nxt, hs + (size_t)rb * H_DIM,
                    cbmh + (size_t)(rb / NC) * CBM_PAD, lane);
    }

    // ---- h1 = tanh(feat @ W1^T + b1) ----
#pragma unroll
    for (int s = 0; s < 2; ++s) {
      v8f acc = {};
#pragma unroll
      for (int kc = 0; kc < 15; ++kc)
        acc = WMMA16(load_a_frag(cur, MLP_INP, kc), W1f[s][kc], acc);
      const int col = (s ? nt1 : nt0) * 16 + (lane & 15);
      const float bb = s ? b1v1 : b1v0;
#pragma unroll
      for (int v = 0; v < 8; ++v)
        h1H[v + mAdd][col] = (half_t)tanhf(acc[v] + bb);
    }
    __syncthreads();

    // ---- h2 = tanh(h1 @ W2^T + b2) ----
#pragma unroll
    for (int s = 0; s < 2; ++s) {
      v8f acc = {};
#pragma unroll
      for (int kc = 0; kc < 8; ++kc)
        acc = WMMA16(load_a_frag(&h1H[0][0], HID, kc), W2f[s][kc], acc);
      const int col = (s ? nt1 : nt0) * 16 + (lane & 15);
      const float bb = s ? b2v1 : b2v0;
#pragma unroll
      for (int v = 0; v < 8; ++v)
        h2H[v + mAdd][col] = (half_t)tanhf(acc[v] + bb);
    }
    __syncthreads();

    // ---- params = h2 @ W3^T + b3 (waves 0..3) ----
    if (wave < 4) {
      v8f acc = {};
#pragma unroll
      for (int kc = 0; kc < 8; ++kc)
        acc = WMMA16(load_a_frag(&h2H[0][0], HID, kc), W3f[kc], acc);
#pragma unroll
      for (int v = 0; v < 8; ++v)
        pLds[v + mAdd][col3] = acc[v] + b3v;
    }
    __syncthreads();

    // ---- mixture log-likelihood, one lane per row ----
    if (tid < MBLK_M) {
      const int grow = rowBase + tid;
      const int bI = grow >> 6, tI = grow & 63;
      const float zt = z[(size_t)bI * NC + tI];
      float maxF = -1e30f, maxL = -1e30f;
      for (int k = 0; k < K_MIX; ++k) {
        const float lo = pLds[tid][k];
        const float mu = pLds[tid][20 + k];
        const float ls = pLds[tid][40 + k];
        const float d  = (zt - mu) * __expf(-ls);
        const float f  = -0.5f * d * d - ls - HALF_L2PI + lo;
        maxF = fmaxf(maxF, f);
        maxL = fmaxf(maxL, lo);
      }
      float sF = 0.0f, sL = 0.0f;
      for (int k = 0; k < K_MIX; ++k) {
        const float lo = pLds[tid][k];
        const float mu = pLds[tid][20 + k];
        const float ls = pLds[tid][40 + k];
        const float d  = (zt - mu) * __expf(-ls);
        const float f  = -0.5f * d * d - ls - HALF_L2PI + lo;
        sF += __expf(f - maxF);
        sL += __expf(lo - maxL);
      }
      const float ll = (maxF + __logf(sF)) - (maxL + __logf(sL));
      llbuf[grow] = (tI < count[bI]) ? ll : 0.0f;
    }
    __syncthreads();
  }
}

// ---------------- deterministic final reduction ----------------

__global__ void reduce_out(const float* __restrict__ llbuf, float* __restrict__ out) {
  const int b = blockIdx.x * blockDim.x + threadIdx.x;
  if (b < B_SZ) {
    float s = 0.0f;
    for (int t = 0; t < NC; ++t) s += llbuf[(size_t)b * NC + t];
    out[b] = s;
  }
}

// ---------------- host launch ----------------

extern "C" void kernel_launch(void* const* d_in, const int* in_sizes, int n_in,
                              void* d_out, int out_size, void* d_ws, size_t ws_size,
                              hipStream_t stream) {
  const float* z    = (const float*)d_in[0];
  const float* c    = (const float*)d_in[1];
  const float* b    = (const float*)d_in[2];
  const float* m    = (const float*)d_in[3];
  const float* W_ih = (const float*)d_in[4];
  const float* W_hh = (const float*)d_in[5];
  const float* b_ih = (const float*)d_in[6];
  const float* b_hh = (const float*)d_in[7];
  const float* W1   = (const float*)d_in[8];
  const float* b1   = (const float*)d_in[9];
  const float* W2   = (const float*)d_in[10];
  const float* b2   = (const float*)d_in[11];
  const float* W3   = (const float*)d_in[12];
  const float* b3   = (const float*)d_in[13];
  float* out = (float*)d_out;

  char* ws = (char*)d_ws;
  size_t off = 0;
  auto take = [&](size_t bytes) -> char* {
    char* p = ws + off;
    off = (off + bytes + 255) & ~(size_t)255;
    return p;
  };
  half_t* cbmh  = (half_t*)take((size_t)B_SZ * CBM_PAD * 2);
  int*    count = (int*)   take((size_t)B_SZ * 4);
  half_t* WihH  = (half_t*)take((size_t)G3H * CBM_PAD * 2);
  half_t* WhhH  = (half_t*)take((size_t)G3H * H_DIM * 2);
  half_t* W1H   = (half_t*)take((size_t)HID * MLP_INP * 2);
  half_t* W2H   = (half_t*)take((size_t)HID * HID * 2);
  half_t* W3H   = (half_t*)take((size_t)P_PAD * H_DIM * 2);
  float*  gi    = (float*) take((size_t)B_SZ * G3H * 4);
  half_t* hs    = (half_t*)take((size_t)B_SZ * NC * H_DIM * 2);   // 128 MB
  float*  llbuf = (float*) take((size_t)B_SZ * NC * 4);           // 1 MB
  (void)ws_size; (void)in_sizes; (void)n_in; (void)out_size;

  prep_rows<<<B_SZ, 256, 0, stream>>>(c, b, m, cbmh, count);

  const int wElems = G3H * CBM_PAD;
  prep_weights<<<(wElems + 255) / 256, 256, 0, stream>>>(
      W_ih, W_hh, W1, W2, W3, WihH, WhhH, W1H, W2H, W3H);

  gi_gemm<<<(B_SZ / 16) * (G3H / 16), 32, 0, stream>>>(cbmh, WihH, b_ih, gi);

  gru_persist<<<B_SZ / MBLK, 512, 0, stream>>>(z, gi, b_hh, W_ih, WhhH, hs);

  mlp_persist<<<256, 256, 0, stream>>>(hs, cbmh, z, count,
                                       W1H, b1, W2H, b2, W3H, b3, llbuf);

  reduce_out<<<(B_SZ + 255) / 256, 256, 0, stream>>>(llbuf, out);
}
